// DecoderLayer_61710090109671
// MI455X (gfx1250) — matrix operand order repaired and checked by reading, not compiled, not run on hardware
//
#include <hip/hip_runtime.h>
#include <hip/hip_bf16.h>
#include <stdint.h>

// Problem constants (match reference)
static constexpr int Bb = 4, Ss = 1024, Dd = 1024, Hh = 16, HDh = 64;

typedef __attribute__((ext_vector_type(16))) __bf16 v16bf;
typedef __attribute__((ext_vector_type(8)))  float  v8f;
typedef __attribute__((ext_vector_type(4)))  unsigned int u32x4;
typedef __attribute__((ext_vector_type(8)))  int i32x8;
typedef __attribute__((ext_vector_type(4)))  int i32x4;

#if __has_builtin(__builtin_amdgcn_tensor_load_to_lds)
#define USE_TDM 1
#if __has_include(<hip/amd_detail/amd_gfx1250_TDM.h>)
#define TDM_ARGS6 1
#else
#define TDM_ARGS6 0
#endif
#else
#define USE_TDM 0
#endif

union FragBF { unsigned int u[8]; v16bf v; };

__device__ __forceinline__ unsigned short f2bf(float f) {
    unsigned int x = __float_as_uint(f);
    return (unsigned short)((x + 0x7FFFu + ((x >> 16) & 1u)) >> 16);
}

__device__ __forceinline__ v8f wmma_bf16(const FragBF& a, const FragBF& b, v8f c) {
    return __builtin_amdgcn_wmma_f32_16x16x32_bf16(false, a.v, false, b.v, (short)0, c,
                                                   false, false);
}

// Compiler barrier: tensor_load_to_lds writes LDS through a descriptor the
// compiler can't alias-analyze; this makes it assume smem was modified so the
// ds_load fragment gathers are not folded away as reads of unwritten memory.
__device__ __forceinline__ void lds_escape(const void* p) {
    asm volatile("" ::"v"(p) : "memory");
}

#if USE_TDM
// Issue one TDM 2D bf16 tile load: tile_h rows x tile_w elems, global row
// stride row_stride (elems), destination LDS byte offset lds_byte_off.
// LDS padding: pad_amount DWORD(s) every 2^(pad_int_code+1) DWORDs.
__device__ __forceinline__ void tdm_load_tile_bf16(
    unsigned lds_byte_off, const unsigned short* gsrc, unsigned tile_w,
    unsigned tile_h, unsigned long long row_stride, unsigned pad_int_code,
    unsigned pad_amt_code, unsigned pad_en) {
    unsigned long long ga = (unsigned long long)gsrc;
    u32x4 g0;
    g0[0] = 1u;  // count=1 (valid user descriptor)
    g0[1] = lds_byte_off;
    g0[2] = (unsigned)(ga & 0xFFFFFFFFu);
    g0[3] = (unsigned)((ga >> 32) & 0x01FFFFFFu) | (2u << 30);  // type=2 (image)
    i32x8 g1;
    // word0: workgroup_mask=0 | data_size=1(2B)@16 | pad_en@20 | pad_interval@22 | pad_amount@25
    g1[0] = (int)((1u << 16) | (pad_en << 20) | (pad_int_code << 22) |
                  (pad_amt_code << 25));
    // tensor_dim0 bits[79:48] = tile_w ; tensor_dim1 bits[111:80] = tile_h
    g1[1] = (int)((tile_w & 0xFFFFu) << 16);
    g1[2] = (int)(((tile_w >> 16) & 0xFFFFu) | ((tile_h & 0xFFFFu) << 16));
    // tile_dim0 bits[127:112]
    g1[3] = (int)(((tile_h >> 16) & 0xFFFFu) | ((tile_w & 0xFFFFu) << 16));
    g1[4] = (int)(tile_h & 0xFFFFu);  // tile_dim1 (tile_dim2 = 0)
    g1[5] = (int)(row_stride & 0xFFFFFFFFull);      // tensor_dim0_stride lo
    g1[6] = (int)((row_stride >> 32) & 0xFFFFull);  // stride hi (dim1_stride=0)
    g1[7] = 0;
    i32x4 z = {};
#if TDM_ARGS6
    i32x8 z8 = {};
    __builtin_amdgcn_tensor_load_to_lds(g0, g1, z, z, z8, 0);
#else
    __builtin_amdgcn_tensor_load_to_lds(g0, g1, z, z, 0);
#endif
}
#endif

// ---------------------------------------------------------------- convert ---
__global__ void cvt_f32_bf16(const float* __restrict__ in, unsigned short* __restrict__ out,
                             long long n) {
    long long i = (long long)blockIdx.x * blockDim.x + threadIdx.x;
    long long stride = (long long)gridDim.x * blockDim.x;
    for (; i < n; i += stride) out[i] = f2bf(in[i]);
}

// Transpose-convert weights: in[K][N] f32 -> out[N][K] bf16 (32x32 tiles).
__global__ __launch_bounds__(256) void cvt_transpose_bf16(
    const float* __restrict__ in, unsigned short* __restrict__ out, int K, int N) {
    __shared__ float t[32][33];
    const int n0 = blockIdx.x * 32, k0 = blockIdx.y * 32;
    const int tx = threadIdx.x & 31, ty = threadIdx.x >> 5;  // 32 x 8
#pragma unroll
    for (int i = 0; i < 32; i += 8)
        t[ty + i][tx] = in[(size_t)(k0 + ty + i) * N + n0 + tx];
    __syncthreads();
#pragma unroll
    for (int i = 0; i < 32; i += 8)
        out[(size_t)(n0 + ty + i) * K + k0 + tx] = f2bf(t[tx][ty + i]);
}

// ------------------------------------------------------------------- GEMM ---
// C[M,N] = A[M,K](bf16) @ Wt[N,K]^T(bf16) + bias, optional ReLU.
// Block: 256 threads (8 waves, 4x2). Block tile 128x128, K-step 32.
// Wave: 32 rows x 64 cols = 2x4 WMMA C tiles (8 WMMAs per K-step).
// TDM path: double-buffered tensor_load_to_lds, one barrier per K-step.
__global__ __launch_bounds__(256) void gemm_bf16_kernel(
    const unsigned short* __restrict__ A, const unsigned short* __restrict__ Wt,
    const float* __restrict__ bias, float* __restrict__ outF,
    unsigned short* __restrict__ outB, int M, int N, int K, int relu) {
    __shared__ unsigned short smem[2 * 2 * 128 * 34];  // 2 buffers x (As|Bs)

    const int tid = threadIdx.x;
    const int w = tid >> 5;
    const int lane = tid & 31;
    const int lo = lane >> 4;
    const int ml = lane & 15;
    const int mw = (w & 3) * 32;   // wave M offset in block tile
    const int nw = (w >> 2) * 64;  // wave N offset in block tile
    const int m0 = blockIdx.y * 128;
    const int n0 = blockIdx.x * 128;

    v8f c[2][4] = {};

#if USE_TDM
    const int nIter = K >> 5;
    if (w == 0) {
        // 128x32 bf16 tiles; LDS rows padded 32->34 (pad 1 DWORD / 16 DWORDs)
        tdm_load_tile_bf16(0u, A + (size_t)m0 * K, 32u, 128u,
                           (unsigned long long)K, 3u, 0u, 1u);
        tdm_load_tile_bf16(128u * 34u * 2u, Wt + (size_t)n0 * K, 32u, 128u,
                           (unsigned long long)K, 3u, 0u, 1u);
    }
    for (int it = 0; it < nIter; ++it) {
        unsigned short* As = smem + (it & 1) * (2 * 128 * 34);
        unsigned short* Bs = As + 128 * 34;
        if (w == 0) __builtin_amdgcn_s_wait_tensorcnt(0);
        __syncthreads();          // publish buffer (it)
        lds_escape(smem);
        if (w == 0 && it + 1 < nIter) {  // prefetch (it+1) into other buffer
            unsigned nb = (unsigned)(((it + 1) & 1) * (2 * 128 * 34) * 2);
            int kk = (it + 1) * 32;
            tdm_load_tile_bf16(nb, A + (size_t)m0 * K + kk, 32u, 128u,
                               (unsigned long long)K, 3u, 0u, 1u);
            tdm_load_tile_bf16(nb + 128u * 34u * 2u, Wt + (size_t)n0 * K + kk, 32u,
                               128u, (unsigned long long)K, 3u, 0u, 1u);
        }
#else
    for (int it = 0, kk = 0; kk < K; ++it, kk += 32) {
        unsigned short* As = smem;
        unsigned short* Bs = smem + 128 * 34;
        {
            int r = tid >> 1, cb = (tid & 1) * 16;
            const uint4* sa = (const uint4*)(A + (size_t)(m0 + r) * K + kk + cb);
            uint4 a0 = sa[0], a1 = sa[1];
            unsigned int* da = (unsigned int*)&As[r * 34 + cb];
            da[0] = a0.x; da[1] = a0.y; da[2] = a0.z; da[3] = a0.w;
            da[4] = a1.x; da[5] = a1.y; da[6] = a1.z; da[7] = a1.w;
            const uint4* sb = (const uint4*)(Wt + (size_t)(n0 + r) * K + kk + cb);
            uint4 b0 = sb[0], b1 = sb[1];
            unsigned int* db = (unsigned int*)&Bs[r * 34 + cb];
            db[0] = b0.x; db[1] = b0.y; db[2] = b0.z; db[3] = b0.w;
            db[4] = b1.x; db[5] = b1.y; db[6] = b1.z; db[7] = b1.w;
        }
        __syncthreads();
#endif

        FragBF fa[2];
#pragma unroll
        for (int mi = 0; mi < 2; ++mi) {
            const int arow = mw + mi * 16 + ml;
#pragma unroll
            for (int v = 0; v < 8; ++v) {
                int k = ((v & 4) ? 16 : 0) + lo * 8 + (v & 3) * 2;
                fa[mi].u[v] = *(const unsigned int*)&As[arow * 34 + k];
            }
        }
#pragma unroll
        for (int nt = 0; nt < 4; ++nt) {
            FragBF fb;
#pragma unroll
            for (int v = 0; v < 8; ++v) {
                int k = ((v & 4) ? 16 : 0) + lo * 8 + (v & 3) * 2;
                fb.u[v] = *(const unsigned int*)&Bs[(nw + nt * 16 + ml) * 34 + k];
            }
            c[0][nt] = wmma_bf16(fa[0], fb, c[0][nt]);
            c[1][nt] = wmma_bf16(fa[1], fb, c[1][nt]);
        }
#if !USE_TDM
        __syncthreads();
#endif
    }

#pragma unroll
    for (int nt = 0; nt < 4; ++nt) {
        int col = n0 + nw + nt * 16 + ml;
        float bv = bias[col];
#pragma unroll
        for (int mi = 0; mi < 2; ++mi) {
#pragma unroll
            for (int r = 0; r < 8; ++r) {
                int row = m0 + mw + mi * 16 + lo * 8 + r;
                float val = c[mi][nt][r] + bv;
                if (relu) val = fmaxf(val, 0.f);
                if (outF) outF[(size_t)row * N + col] = val;
                if (outB) outB[(size_t)row * N + col] = f2bf(val);
            }
        }
    }
}

// -------------------------------------------------------- flash attention ---
// grid: (S/64, B*H). Block: 128 threads (4 waves); wave -> 16 query rows.
__global__ __launch_bounds__(128) void attn_kernel(
    const unsigned short* __restrict__ Qp, int q_stride, int q_base,
    const unsigned short* __restrict__ Kp, const unsigned short* __restrict__ Vp,
    int kv_stride, int k_base, int v_base, float* __restrict__ outp,
    const int* __restrict__ Lq, const int* __restrict__ Lkv, int causal) {
    // Ks: [32 keys][64 hd] stride 66 | Vt: [64 hd][32 keys] stride 34 | Ps: 4x(16x34)
    __shared__ unsigned short smem[32 * 66 + 64 * 34 + 4 * 16 * 34];
    unsigned short* Ks = smem;
    unsigned short* Vt = smem + 32 * 66;
    unsigned short* Ps = smem + 32 * 66 + 64 * 34;

    const int tid = threadIdx.x;
    const int w = tid >> 5, lane = tid & 31, lo = lane >> 4, ml = lane & 15;
    const int bh = (int)blockIdx.y;
    const int b = bh / Hh, h = bh % Hh;
    const int qblk = (int)blockIdx.x * 64;
    const int lq = Lq[b], lkv = Lkv[b];
    const int hoff = h * HDh;
    const float scale = 1.0f / 32.0f;  // 1/sqrt(D), D=1024

    FragBF q0, q1;
    {
        const int row = qblk + w * 16 + ml;
        const unsigned short* qr = Qp + (size_t)(b * Ss + row) * q_stride + q_base + hoff;
#pragma unroll
        for (int v = 0; v < 8; ++v) {
            int k = ((v & 4) ? 16 : 0) + lo * 8 + (v & 3) * 2;
            q0.u[v] = *(const unsigned int*)&qr[k];
            q1.u[v] = *(const unsigned int*)&qr[32 + k];
        }
    }

    v8f o[4] = {};
    float mrun[8], lrun[8];
#pragma unroll
    for (int r = 0; r < 8; ++r) { mrun[r] = -1e30f; lrun[r] = 0.f; }

    int nkeys = causal ? (qblk + 64) : Ss;
    int lim = (lkv + 31) & ~31;
    if (lim < nkeys) nkeys = lim;
    if (nkeys > Ss) nkeys = Ss;

    for (int kb = 0; kb < nkeys; kb += 32) {
#if USE_TDM
        if (w == 0) {
            // 32x64 bf16 K tile; LDS rows padded 64->66 (1 DWORD / 32 DWORDs)
            tdm_load_tile_bf16(
                0u, Kp + (size_t)(b * Ss + kb) * kv_stride + k_base + hoff, 64u, 32u,
                (unsigned long long)kv_stride, 4u, 0u, 1u);
        }
#else
        {
            int key = tid >> 2, cb = (tid & 3) * 16;
            const unsigned short* kr =
                Kp + (size_t)(b * Ss + kb + key) * kv_stride + k_base + hoff + cb;
            uint4 d0 = *(const uint4*)kr;
            uint4 d1 = *(const uint4*)(kr + 8);
            unsigned int* dk = (unsigned int*)&Ks[key * 66 + cb];
            dk[0] = d0.x; dk[1] = d0.y; dk[2] = d0.z; dk[3] = d0.w;
            dk[4] = d1.x; dk[5] = d1.y; dk[6] = d1.z; dk[7] = d1.w;
        }
#endif
        {   // V chunk staged transposed (manual) -- overlaps the K-tile DMA
            int key = tid >> 2, cb = (tid & 3) * 16;
            const unsigned short* vr =
                Vp + (size_t)(b * Ss + kb + key) * kv_stride + v_base + hoff + cb;
            uint4 e0 = *(const uint4*)vr;
            uint4 e1 = *(const uint4*)(vr + 8);
            unsigned int ev[8] = {e0.x, e0.y, e0.z, e0.w, e1.x, e1.y, e1.z, e1.w};
#pragma unroll
            for (int i = 0; i < 8; ++i) {
                Vt[(cb + 2 * i + 0) * 34 + key] = (unsigned short)(ev[i] & 0xFFFFu);
                Vt[(cb + 2 * i + 1) * 34 + key] = (unsigned short)(ev[i] >> 16);
            }
        }
#if USE_TDM
        if (w == 0) __builtin_amdgcn_s_wait_tensorcnt(0);
#endif
        __syncthreads();
        lds_escape(smem);

        v8f s0 = {}, s1 = {};
        {
            FragBF fb0, fb1;
#pragma unroll
            for (int v = 0; v < 8; ++v) {
                int k = ((v & 4) ? 16 : 0) + lo * 8 + (v & 3) * 2;
                fb0.u[v] = *(const unsigned int*)&Ks[ml * 66 + k];
                fb1.u[v] = *(const unsigned int*)&Ks[ml * 66 + 32 + k];
            }
            s0 = wmma_bf16(q0, fb0, s0);
            s0 = wmma_bf16(q1, fb1, s0);
#pragma unroll
            for (int v = 0; v < 8; ++v) {
                int k = ((v & 4) ? 16 : 0) + lo * 8 + (v & 3) * 2;
                fb0.u[v] = *(const unsigned int*)&Ks[(16 + ml) * 66 + k];
                fb1.u[v] = *(const unsigned int*)&Ks[(16 + ml) * 66 + 32 + k];
            }
            s1 = wmma_bf16(q0, fb0, s1);
            s1 = wmma_bf16(q1, fb1, s1);
        }

        float p0[8], p1[8], alpha[8];
        {
            int key0 = kb + ml, key1 = kb + 16 + ml;
#pragma unroll
            for (int r = 0; r < 8; ++r) {
                int row = qblk + w * 16 + lo * 8 + r;
                bool rv = (row < lq);
                bool v0 = rv && (key0 < lkv) && (!causal || key0 <= row);
                bool v1 = rv && (key1 < lkv) && (!causal || key1 <= row);
                float a0 = v0 ? s0[r] * scale : -1e30f;
                float a1 = v1 ? s1[r] * scale : -1e30f;
                float t = fmaxf(a0, a1);
                t = fmaxf(t, __shfl_xor(t, 1));
                t = fmaxf(t, __shfl_xor(t, 2));
                t = fmaxf(t, __shfl_xor(t, 4));
                t = fmaxf(t, __shfl_xor(t, 8));
                float mnew = fmaxf(mrun[r], t);
                alpha[r] = __expf(mrun[r] - mnew);
                mrun[r] = mnew;
                p0[r] = v0 ? __expf(a0 - mnew) : 0.f;
                p1[r] = v1 ? __expf(a1 - mnew) : 0.f;
                float rs = p0[r] + p1[r];
                rs += __shfl_xor(rs, 1);
                rs += __shfl_xor(rs, 2);
                rs += __shfl_xor(rs, 4);
                rs += __shfl_xor(rs, 8);
                lrun[r] = lrun[r] * alpha[r] + rs;
            }
#pragma unroll
            for (int nt = 0; nt < 4; ++nt)
#pragma unroll
                for (int r = 0; r < 8; ++r) o[nt][r] *= alpha[r];
        }

        {
            unsigned short* ps = &Ps[w * 16 * 34];
#pragma unroll
            for (int r = 0; r < 8; ++r) {
                ps[(lo * 8 + r) * 34 + ml] = f2bf(p0[r]);
                ps[(lo * 8 + r) * 34 + 16 + ml] = f2bf(p1[r]);
            }
            asm volatile("s_wait_dscnt 0" ::: "memory");
            FragBF pf;
#pragma unroll
            for (int v = 0; v < 8; ++v) {
                int k = ((v & 4) ? 16 : 0) + lo * 8 + (v & 3) * 2;
                pf.u[v] = *(const unsigned int*)&ps[ml * 34 + k];
            }
#pragma unroll
            for (int nt = 0; nt < 4; ++nt) {
                FragBF vb;
#pragma unroll
                for (int v = 0; v < 8; ++v) {
                    int k = ((v & 4) ? 16 : 0) + lo * 8 + (v & 3) * 2;
                    vb.u[v] = *(const unsigned int*)&Vt[(nt * 16 + ml) * 34 + k];
                }
                o[nt] = wmma_bf16(pf, vb, o[nt]);
            }
        }
        __syncthreads();
    }

    float inv[8];
#pragma unroll
    for (int r = 0; r < 8; ++r) inv[r] = (lrun[r] > 0.f) ? (1.f / lrun[r]) : 0.f;
#pragma unroll
    for (int nt = 0; nt < 4; ++nt) {
        int col = hoff + nt * 16 + ml;
#pragma unroll
        for (int r = 0; r < 8; ++r) {
            int row = qblk + w * 16 + lo * 8 + r;
            outp[(size_t)(b * Ss + row) * Dd + col] = o[nt][r] * inv[r];
        }
    }
}

// ----------------------------------------------------- residual + layernorm ---
__global__ __launch_bounds__(256) void ln_res_kernel(
    const float* __restrict__ x, const float* __restrict__ rc,
    const float* __restrict__ g, const float* __restrict__ be,
    float* __restrict__ outF, unsigned short* __restrict__ outB) {
    __shared__ float s1[256], s2[256];
    const int row = blockIdx.x, t = threadIdx.x;
    const float* xr = x + (size_t)row * Dd;
    const float* rr = rc + (size_t)row * Dd;
    float v[4];
    float a = 0.f, q = 0.f;
#pragma unroll
    for (int i = 0; i < 4; ++i) {
        int c = t + i * 256;
        v[i] = xr[c] + rr[c];
        a += v[i];
        q += v[i] * v[i];
    }
    s1[t] = a; s2[t] = q;
    __syncthreads();
    for (int s = 128; s > 0; s >>= 1) {
        if (t < s) { s1[t] += s1[t + s]; s2[t] += s2[t + s]; }
        __syncthreads();
    }
    float mu = s1[0] * (1.0f / Dd);
    float var = s2[0] * (1.0f / Dd) - mu * mu;
    float inv = rsqrtf(var + 1e-5f);
#pragma unroll
    for (int i = 0; i < 4; ++i) {
        int c = t + i * 256;
        float y = (v[i] - mu) * inv * g[c] + be[c];
        outF[(size_t)row * Dd + c] = y;
        if (outB) outB[(size_t)row * Dd + c] = f2bf(y);
    }
}

// ------------------------------------------------------------------ driver ---
extern "C" void kernel_launch(void* const* d_in, const int* in_sizes, int n_in,
                              void* d_out, int out_size, void* d_ws, size_t ws_size,
                              hipStream_t stream) {
    (void)in_sizes; (void)n_in; (void)out_size; (void)ws_size;

    const float* en     = (const float*)d_in[0];
    const float* fr     = (const float*)d_in[1];
    const float* W_attn = (const float*)d_in[2];
    const float* b_attn = (const float*)d_in[3];
    const float* W_Q    = (const float*)d_in[4];
    const float* b_Q    = (const float*)d_in[5];
    const float* W_KV   = (const float*)d_in[6];
    const float* b_KV   = (const float*)d_in[7];
    const float* ln_g   = (const float*)d_in[8];
    const float* ln_b   = (const float*)d_in[9];
    const float* W1     = (const float*)d_in[10];
    const float* b1     = (const float*)d_in[11];
    const float* W2     = (const float*)d_in[12];
    const float* b2     = (const float*)d_in[13];
    const int*   l_en   = (const int*)d_in[14];
    const int*   l_fr   = (const int*)d_in[15];
    float* out = (float*)d_out;

    const size_t BSD = (size_t)Bb * Ss * Dd;
    const int rows = Bb * Ss;  // 4096

    char* ws = (char*)d_ws;
    size_t off = 0;
    auto alloc = [&](size_t bytes) -> void* {
        void* p = ws + off;
        off = (off + bytes + 255) & ~(size_t)255;
        return p;
    };

    unsigned short* fr_bf    = (unsigned short*)alloc(BSD * 2);
    unsigned short* en_bf    = (unsigned short*)alloc(BSD * 2);
    unsigned short* Wattn_t  = (unsigned short*)alloc((size_t)Dd * 3 * Dd * 2);  // [3D,D]
    unsigned short* WQ_t     = (unsigned short*)alloc((size_t)Dd * Dd * 2);      // [D,D]
    unsigned short* WKV_t    = (unsigned short*)alloc((size_t)Dd * 2 * Dd * 2);  // [2D,D]
    unsigned short* W1_t     = (unsigned short*)alloc((size_t)Dd * 4 * Dd * 2);  // [4D,D]
    unsigned short* W2_t     = (unsigned short*)alloc((size_t)4 * Dd * Dd * 2);  // [D,4D]
    unsigned short* qkv_bf   = (unsigned short*)alloc((size_t)rows * 3 * Dd * 2);
    unsigned short* q_bf     = (unsigned short*)alloc(BSD * 2);
    unsigned short* kv_bf    = (unsigned short*)alloc((size_t)rows * 2 * Dd * 2);
    float* xbuf              = (float*)alloc(BSD * 4);
    float* fr2f              = (float*)alloc(BSD * 4);
    unsigned short* fr2bf    = (unsigned short*)alloc(BSD * 2);
    float* fr3f              = (float*)alloc(BSD * 4);
    unsigned short* fr3bf    = (unsigned short*)alloc(BSD * 2);
    unsigned short* h1_bf = qkv_bf;  // 32MB aliases qkv(24MB)+q(8MB), dead by FFN

    auto cvt = [&](const float* src, unsigned short* dst, size_t n) {
        cvt_f32_bf16<<<2048, 256, 0, stream>>>(src, dst, (long long)n);
    };
    auto cvtT = [&](const float* src, unsigned short* dst, int K, int N) {
        cvt_transpose_bf16<<<dim3(N / 32, K / 32), 256, 0, stream>>>(src, dst, K, N);
    };

    // 1) convert activations; transpose-convert weights (TDM-friendly [N,K])
    cvt(fr, fr_bf, BSD);
    cvt(en, en_bf, BSD);
    cvtT(W_attn, Wattn_t, Dd, 3 * Dd);
    cvtT(W_Q, WQ_t, Dd, Dd);
    cvtT(W_KV, WKV_t, Dd, 2 * Dd);
    cvtT(W1, W1_t, Dd, 4 * Dd);
    cvtT(W2, W2_t, 4 * Dd, Dd);

    // 2) fused QKV projection
    gemm_bf16_kernel<<<dim3(3 * Dd / 128, rows / 128), 256, 0, stream>>>(
        fr_bf, Wattn_t, b_attn, nullptr, qkv_bf, rows, 3 * Dd, Dd, 0);

    // 3) causal + pad masked self-attention
    attn_kernel<<<dim3(Ss / 64, Bb * Hh), 128, 0, stream>>>(
        qkv_bf, 3 * Dd, 0, qkv_bf, qkv_bf, 3 * Dd, Dd, 2 * Dd, xbuf, l_fr, l_fr, 1);

    // 4) residual + LN -> fr2
    ln_res_kernel<<<rows, 256, 0, stream>>>(xbuf, fr, ln_g, ln_b, fr2f, fr2bf);

    // 5) cross-attn projections
    gemm_bf16_kernel<<<dim3(Dd / 128, rows / 128), 256, 0, stream>>>(
        fr2bf, WQ_t, b_Q, nullptr, q_bf, rows, Dd, Dd, 0);
    gemm_bf16_kernel<<<dim3(2 * Dd / 128, rows / 128), 256, 0, stream>>>(
        en_bf, WKV_t, b_KV, nullptr, kv_bf, rows, 2 * Dd, Dd, 0);

    // 6) cross attention (pad mask only)
    attn_kernel<<<dim3(Ss / 64, Bb * Hh), 128, 0, stream>>>(
        q_bf, Dd, 0, kv_bf, kv_bf, 2 * Dd, 0, Dd, xbuf, l_fr, l_en, 0);

    // 7) residual + LN -> fr3
    ln_res_kernel<<<rows, 256, 0, stream>>>(xbuf, fr2f, ln_g, ln_b, fr3f, fr3bf);

    // 8) FFN up (ReLU)
    gemm_bf16_kernel<<<dim3(4 * Dd / 128, rows / 128), 256, 0, stream>>>(
        fr3bf, W1_t, b1, nullptr, h1_bf, rows, 4 * Dd, Dd, 1);

    // 9) FFN down + extra ReLU
    gemm_bf16_kernel<<<dim3(Dd / 128, rows / 128), 256, 0, stream>>>(
        h1_bf, W2_t, b2, xbuf, nullptr, rows, Dd, 4 * Dd, 1);

    // 10) residual + LN -> final output
    ln_res_kernel<<<rows, 256, 0, stream>>>(xbuf, fr3f, ln_g, ln_b, out, nullptr);
}
